// BitLinear158_27917287424034
// MI455X (gfx1250) — compile-verified, matrix operand order checked by reading, not run on hardware
//
#include <hip/hip_runtime.h>

// Problem dims (fixed by reference): x[4,2048,2048] f32, w[8192,2048] f32, bias[8192]
#define IN_DIM   2048
#define OUT_DIM  8192
#define M_DIM    8192                    // 4 * 2048
#define W_ELEMS  (OUT_DIM * IN_DIM)      // 16777216
#define X_ELEMS  (M_DIM * IN_DIM)        // 16777216

typedef __attribute__((ext_vector_type(16))) __bf16 v16bf;
typedef __attribute__((ext_vector_type(8)))  __bf16 v8bf;
typedef __attribute__((ext_vector_type(8)))  float  v8f;
typedef __attribute__((ext_vector_type(4)))  float  v4f;

union Frag16 { v16bf v; v8bf h[2]; };

// ---------------------------------------------------------------------------
// Kernel 1: zero the reduction accumulator (graph-capture-safe, no memset)
// ---------------------------------------------------------------------------
__global__ void zero_sum_kernel(float* __restrict__ sum) {
    if (threadIdx.x == 0 && blockIdx.x == 0) sum[0] = 0.0f;
}

// ---------------------------------------------------------------------------
// Kernel 2: sum(|w|) -> ws[0]  (grid-stride, block reduce, f32 atomicAdd)
// ---------------------------------------------------------------------------
__global__ __launch_bounds__(256) void absmean_kernel(const float* __restrict__ w,
                                                      float* __restrict__ sum) {
    const int tid    = blockIdx.x * 256 + threadIdx.x;
    const int stride = gridDim.x * 256;
    const v4f* w4 = (const v4f*)w;
    const int n4  = W_ELEMS / 4;

    float acc = 0.0f;
    for (int i = tid; i < n4; i += stride) {
        v4f v = w4[i];
        acc += __builtin_fabsf(v.x) + __builtin_fabsf(v.y) +
               __builtin_fabsf(v.z) + __builtin_fabsf(v.w);
    }

    __shared__ float red[256];
    red[threadIdx.x] = acc;
    __syncthreads();
    #pragma unroll
    for (int s = 128; s > 0; s >>= 1) {
        if (threadIdx.x < s) red[threadIdx.x] += red[threadIdx.x + s];
        __syncthreads();
    }
    if (threadIdx.x == 0) atomicAdd(sum, red[0]);
}

// ---------------------------------------------------------------------------
// Kernel 3: x (f32) -> x_bf16.  8 elems/thread, 16B bf16 stores.
// ---------------------------------------------------------------------------
__global__ __launch_bounds__(256) void convx_kernel(const float* __restrict__ x,
                                                    __bf16* __restrict__ xb) {
    const int t = blockIdx.x * 256 + threadIdx.x;   // 0 .. X_ELEMS/8 - 1
    const v4f* x4 = (const v4f*)x;
    v4f a = x4[2 * t + 0];
    v4f b = x4[2 * t + 1];
    v8bf o;
    o[0] = (__bf16)a.x; o[1] = (__bf16)a.y; o[2] = (__bf16)a.z; o[3] = (__bf16)a.w;
    o[4] = (__bf16)b.x; o[5] = (__bf16)b.y; o[6] = (__bf16)b.z; o[7] = (__bf16)b.w;
    ((v8bf*)xb)[t] = o;
}

// ---------------------------------------------------------------------------
// Kernel 4: w (f32) -> ternary bf16 {-1,0,1}.  scale = 1e-5 + sum/NW.
// rintf == round-to-nearest-even, matching jnp.round.  Ternary is exact in bf16.
// ---------------------------------------------------------------------------
__global__ __launch_bounds__(256) void quantw_kernel(const float* __restrict__ w,
                                                     const float* __restrict__ sum,
                                                     __bf16* __restrict__ qw) {
    const float scale = 1e-5f + sum[0] * (1.0f / (float)W_ELEMS);
    const float inv   = 1.0f / scale;
    const int t = blockIdx.x * 256 + threadIdx.x;   // 0 .. W_ELEMS/8 - 1
    const v4f* w4 = (const v4f*)w;
    v4f a = w4[2 * t + 0];
    v4f b = w4[2 * t + 1];
    float f[8] = {a.x, a.y, a.z, a.w, b.x, b.y, b.z, b.w};
    v8bf o;
    #pragma unroll
    for (int i = 0; i < 8; ++i) {
        float q = __builtin_rintf(f[i] * inv);
        q = fminf(1.0f, fmaxf(-1.0f, q));
        o[i] = (__bf16)q;
    }
    ((v8bf*)qw)[t] = o;
}

// ---------------------------------------------------------------------------
// Kernel 5: GEMM  out[M,N] = Xb[M,K] * Qw[N,K]^T + bias[N]
// Block = 256 threads = 8 wave32s; block tile 128(M) x 128(N).
// Wave grid 2(M) x 4(N): each wave owns 64x32 = 4x2 tiles of 16x16.
//
// Software-pipelined: two explicit fragment buffers; loads for K-chunk k+32
// are issued before the WMMAs of chunk k so ~12 b128 loads are always in
// flight behind the matrix pipe.  ~180 VGPRs/wave (64 acc + 2x48 frags).
//
// Fragment layout (16-bit A/B, ISA 7.12.2): lane L holds row (L&15);
// lanes 0-15 take K [0..7]+[16..23], lanes 16-31 take K [8..15]+[24..31]
// -> two aligned 16B global_load_b128 per fragment per lane.  Qw is [N,K]
// row-major (already B^T), so A and B loads are symmetric.
// ---------------------------------------------------------------------------
__device__ __forceinline__ void load_frags(const __bf16* __restrict__ pA,
                                           const __bf16* __restrict__ pB,
                                           int kk, Frag16 a[4], Frag16 b[2]) {
    #pragma unroll
    for (int i = 0; i < 4; ++i) {
        const __bf16* p = pA + (size_t)i * 16 * IN_DIM + kk;
        a[i].h[0] = *(const v8bf*)(p);
        a[i].h[1] = *(const v8bf*)(p + 16);
    }
    #pragma unroll
    for (int j = 0; j < 2; ++j) {
        const __bf16* p = pB + (size_t)j * 16 * IN_DIM + kk;
        b[j].h[0] = *(const v8bf*)(p);
        b[j].h[1] = *(const v8bf*)(p + 16);
    }
}

__device__ __forceinline__ void do_mmas(const Frag16 a[4], const Frag16 b[2],
                                        v8f acc[4][2]) {
    #pragma unroll
    for (int i = 0; i < 4; ++i)
        #pragma unroll
        for (int j = 0; j < 2; ++j)
            acc[i][j] = __builtin_amdgcn_wmma_f32_16x16x32_bf16(
                /*neg_a=*/false, a[i].v,
                /*neg_b=*/false, b[j].v,
                /*c_mod=*/(short)0, acc[i][j],
                /*reuse_a=*/false, /*reuse_b=*/false);
}

__global__ __launch_bounds__(256, 1) void gemm_bf16_kernel(const __bf16* __restrict__ X,
                                                           const __bf16* __restrict__ Wq,
                                                           const float* __restrict__ bias,
                                                           float* __restrict__ out) {
    const int lane = threadIdx.x & 31;
    const int wave = threadIdx.x >> 5;
    const int wm   = wave & 1;        // 0..1 over M
    const int wn   = wave >> 1;       // 0..3 over N

    const int m0 = blockIdx.y * 128 + wm * 64;   // wave M base (64 rows)
    const int n0 = blockIdx.x * 128 + wn * 32;   // wave N base (32 cols)

    const int lrow  = lane & 15;
    const int khalf = (lane >> 4) << 3;          // 0 or 8

    v8f acc[4][2];
    #pragma unroll
    for (int i = 0; i < 4; ++i)
        #pragma unroll
        for (int j = 0; j < 2; ++j)
            acc[i][j] = (v8f)(0.0f);

    const __bf16* pA = X  + (size_t)(m0 + lrow) * IN_DIM + khalf;
    const __bf16* pB = Wq + (size_t)(n0 + lrow) * IN_DIM + khalf;

    Frag16 aE[4], bE[2];   // "even" buffer: chunks kk
    Frag16 aO[4], bO[2];   // "odd"  buffer: chunks kk+32

    load_frags(pA, pB, 0, aE, bE);               // prologue

    for (int kk = 0; kk < IN_DIM; kk += 64) {
        load_frags(pA, pB, kk + 32, aO, bO);     // prefetch odd chunk
        do_mmas(aE, bE, acc);                    // compute even chunk
        if (kk + 64 < IN_DIM)
            load_frags(pA, pB, kk + 64, aE, bE); // prefetch next even chunk
        do_mmas(aO, bO, acc);                    // compute odd chunk
    }

    // Epilogue.  C layout: VGPR r -> M = r (+8 for lanes 16-31), N = lane&15.
    const int rofs = (lane >> 4) << 3;           // 0 or 8
    #pragma unroll
    for (int j = 0; j < 2; ++j) {
        const int col = n0 + j * 16 + lrow;
        const float bv = bias[col];
        #pragma unroll
        for (int i = 0; i < 4; ++i) {
            const int rbase = m0 + i * 16 + rofs;
            #pragma unroll
            for (int r = 0; r < 8; ++r)
                out[(size_t)(rbase + r) * OUT_DIM + col] = acc[i][j][r] + bv;
        }
    }
}

// ---------------------------------------------------------------------------
// Launch
// ---------------------------------------------------------------------------
extern "C" void kernel_launch(void* const* d_in, const int* in_sizes, int n_in,
                              void* d_out, int out_size, void* d_ws, size_t ws_size,
                              hipStream_t stream) {
    const float* x    = (const float*)d_in[0];
    const float* w    = (const float*)d_in[1];
    const float* bias = (const float*)d_in[2];
    float* out        = (float*)d_out;

    // Workspace layout: [0,256) reduction scalar; then x_bf16; then qw_bf16.
    char* ws = (char*)d_ws;
    float*  sum = (float*)ws;
    __bf16* xb  = (__bf16*)(ws + 256);
    __bf16* qw  = (__bf16*)(ws + 256 + (size_t)X_ELEMS * sizeof(__bf16));

    zero_sum_kernel<<<1, 1, 0, stream>>>(sum);
    absmean_kernel<<<1024, 256, 0, stream>>>(w, sum);
    convx_kernel<<<X_ELEMS / 8 / 256, 256, 0, stream>>>(x, xb);
    quantw_kernel<<<W_ELEMS / 8 / 256, 256, 0, stream>>>(w, sum, qw);

    dim3 grid(OUT_DIM / 128, M_DIM / 128);   // (64, 64)
    gemm_bf16_kernel<<<grid, 256, 0, stream>>>(xb, qw, bias, out);
}
